// CrossAttention_29463475650625
// MI455X (gfx1250) — matrix operand order repaired and checked by reading, not compiled, not run on hardware
//
#include <hip/hip_runtime.h>
#include <hip/hip_bf16.h>

typedef __attribute__((ext_vector_type(16))) _Float16 v16h;
typedef __attribute__((ext_vector_type(8)))  _Float16 v8h;
typedef __attribute__((ext_vector_type(8)))  float    v8f;

#define BATCH 8
#define NHEAD 8
#define SEQ   1024
#define CIN   256
#define HD    64
#define NPROJ 512   // NH*HD

__device__ __forceinline__ v8f wmma_f16(v16h a, v16h b, v8f c) {
    // D = A(16x32 f16) * B(32x16 f16) + C(16x16 f32)
    return __builtin_amdgcn_wmma_f32_16x16x32_f16(
        false, a, false, b, (short)0, c, false, false);
}

// ---------------------------------------------------------------------------
// Stage 1: projection GEMM.
//   A[m=s, k=c] = in[b][c][s]           (input is [B, C, S] channel-major)
//   B[k=c, n]   = W[n][c]               (nn.Linear weight [out,in])
// TRANS_OUT == false: out[b][h][s][d]   (for K1 / K2: d-contiguous)
// TRANS_OUT == true : out[b][h][d][s]   (for V2: s-contiguous, so attention
//                                        B-fragments are single v16h loads)
// One wave computes a 16(M) x 64(N) tile; K-loop over 256 in steps of 32.
// ---------------------------------------------------------------------------
template <bool TRANS_OUT>
__global__ __launch_bounds__(256)
void proj_gemm_kernel(const float* __restrict__ in,
                      const float* __restrict__ W,
                      _Float16*    __restrict__ out)
{
    const int lane = threadIdx.x & 31;
    const int wid  = blockIdx.x * 8 + (threadIdx.x >> 5);

    const int b  = wid >> 9;            // 512 wave-tasks per batch
    const int r  = wid & 511;
    const int s0 = (r >> 3) << 4;       // 64 m-tiles of 16
    const int n0 = (r & 7)  << 6;       // 8 n-strips of 64

    const int m   = lane & 15;          // A row
    const int kb  = (lane >> 4) << 3;   // unused (A and B use kb2 {0,16})
    const int n   = lane & 15;          // B col
    const int kb2 = (lane >> 4) << 4;   // B k base {0,16}

    const float* inb = in + (size_t)b * CIN * SEQ;

    v8f acc[4] = {};

    for (int k0 = 0; k0 < CIN; k0 += 32) {
        // A fragment: 16 coalesced-across-lanes f32 loads, convert to f16
        v16h a;
#pragma unroll
        for (int i = 0; i < 16; ++i) {
            int k = k0 + kb2 + i;
            a[i] = (_Float16)inb[k * SEQ + s0 + m];
        }
#pragma unroll
        for (int t = 0; t < 4; ++t) {
            const float* wrow = W + (size_t)(n0 + 16 * t + n) * CIN + k0 + kb2;
            v16h bf;
#pragma unroll
            for (int i = 0; i < 16; ++i) bf[i] = (_Float16)wrow[i];
            acc[t] = wmma_f16(a, bf, acc[t]);
        }
    }

#pragma unroll
    for (int t = 0; t < 4; ++t) {
#pragma unroll
        for (int rr = 0; rr < 8; ++rr) {
            int mm = rr + ((lane >> 4) << 3);
            int nf = n0 + 16 * t + n;
            int h  = nf >> 6, d = nf & 63;
            _Float16 v = (_Float16)acc[t][rr];
            if (TRANS_OUT) {
                // [B, NH, HD, SEQ]
                out[((size_t)(b * NHEAD + h) * HD + d) * SEQ + s0 + mm] = v;
            } else {
                // [B, NH, SEQ, HD]
                out[((size_t)(b * NHEAD + h) * SEQ + s0 + mm) * HD + d] = v;
            }
        }
    }
}

// ---------------------------------------------------------------------------
// Stage 2: flash-style attention.  One wave owns a 16-row q tile of one
// (b, h).  Streams j in steps of 32 with online softmax; P tile goes through
// LDS to re-layout C-matrix form -> A-fragment form for the P*V WMMA.
// V2 is stored transposed [B,H,D,S] so every V B-fragment is one v16h load.
// ---------------------------------------------------------------------------
__global__ __launch_bounds__(256)
void attn_kernel(const _Float16* __restrict__ K1,
                 const _Float16* __restrict__ K2,
                 const _Float16* __restrict__ V2T,
                 _Float16*       __restrict__ O)
{
    __shared__ _Float16 Pbuf[8][16 * 32];   // per-wave P staging (8 KB)

    const int wslot = threadIdx.x >> 5;
    const int lane  = threadIdx.x & 31;
    const int wid   = blockIdx.x * 8 + wslot;

    const int b  = wid >> 9;
    const int r  = wid & 511;
    const int h  = r >> 6;
    const int q0 = (r & 63) << 4;

    const int m     = lane & 15;
    const int kb    = (lane >> 4) << 3;
    const int n     = lane & 15;
    const int kb2   = (lane >> 4) << 4;
    const int mhalf = (lane >> 4) << 3;

    const _Float16* k1row = K1 + ((size_t)(b * NHEAD + h) * SEQ + q0 + m) * HD;
    const _Float16* k2p   = K2  + (size_t)(b * NHEAD + h) * SEQ * HD;
    const _Float16* v2tp  = V2T + (size_t)(b * NHEAD + h) * HD * SEQ;

    // Q-tile A fragments (d = 0..31 and 32..63), halves are two 16B chunks
    v16h a0, a1;
    {
        v8h l0 = *(const v8h*)(k1row + kb2);
        v8h l1 = *(const v8h*)(k1row + kb2 + 8);
        v8h l2 = *(const v8h*)(k1row + 32 + kb2);
        v8h l3 = *(const v8h*)(k1row + 32 + kb2 + 8);
#pragma unroll
        for (int i = 0; i < 8; ++i) {
            a0[i] = l0[i]; a0[i + 8] = l1[i];
            a1[i] = l2[i]; a1[i + 8] = l3[i];
        }
    }

    float mrow[8], lrow[8];
#pragma unroll
    for (int rr = 0; rr < 8; ++rr) { mrow[rr] = -1e30f; lrow[rr] = 0.f; }
    v8f o[4] = {};

    _Float16* Prow = &Pbuf[wslot][0];

    for (int j0 = 0; j0 < SEQ; j0 += 32) {
        // scores for 32 keys: two 16x16 tiles, K-dim = head dim (2 steps)
        v8f s0 = {}, s1 = {};
        {
            v16h b00 = *(const v16h*)(k2p + (size_t)(j0 + n) * HD + kb2);
            v16h b01 = *(const v16h*)(k2p + (size_t)(j0 + n) * HD + 32 + kb2);
            s0 = wmma_f16(a0, b00, s0);
            s0 = wmma_f16(a1, b01, s0);
            v16h b10 = *(const v16h*)(k2p + (size_t)(j0 + 16 + n) * HD + kb2);
            v16h b11 = *(const v16h*)(k2p + (size_t)(j0 + 16 + n) * HD + 32 + kb2);
            s1 = wmma_f16(a0, b10, s1);
            s1 = wmma_f16(a1, b11, s1);
        }

        // online softmax update (row = striped across the 16-lane half)
        float corr[8];
#pragma unroll
        for (int rr = 0; rr < 8; ++rr) {
            float x0 = s0[rr] * 0.125f;     // 1/sqrt(64)
            float x1 = s1[rr] * 0.125f;
            float mx = fmaxf(x0, x1);
            mx = fmaxf(mx, __shfl_xor(mx, 1, 32));
            mx = fmaxf(mx, __shfl_xor(mx, 2, 32));
            mx = fmaxf(mx, __shfl_xor(mx, 4, 32));
            mx = fmaxf(mx, __shfl_xor(mx, 8, 32));
            float mnew = fmaxf(mrow[rr], mx);
            float c  = __expf(mrow[rr] - mnew);
            float p0 = __expf(x0 - mnew);
            float p1 = __expf(x1 - mnew);
            float sum = p0 + p1;
            sum += __shfl_xor(sum, 1, 32);
            sum += __shfl_xor(sum, 2, 32);
            sum += __shfl_xor(sum, 4, 32);
            sum += __shfl_xor(sum, 8, 32);
            lrow[rr] = lrow[rr] * c + sum;
            mrow[rr] = mnew;
            corr[rr] = c;
            int mm = rr + mhalf;
            Prow[mm * 32 + n]      = (_Float16)p0;
            Prow[mm * 32 + 16 + n] = (_Float16)p1;
        }
#pragma unroll
        for (int t = 0; t < 4; ++t)
#pragma unroll
            for (int rr = 0; rr < 8; ++rr) o[t][rr] *= corr[rr];

        __syncthreads();
        // P as A fragment (16x32): two contiguous 16B LDS reads per lane
        v16h pa;
        {
            const _Float16* pr = Prow + m * 32;
            v8h plo = *(const v8h*)(pr + kb2);
            v8h phi = *(const v8h*)(pr + kb2 + 8);
#pragma unroll
            for (int i = 0; i < 8; ++i) { pa[i] = plo[i]; pa[i + 8] = phi[i]; }
        }
        __syncthreads();

        // O += P(16x32) * V(32x64); V fragment = one contiguous v16h load
#pragma unroll
        for (int t = 0; t < 4; ++t) {
            v16h bv = *(const v16h*)(v2tp + (size_t)(16 * t + n) * SEQ + j0 + kb2);
            o[t] = wmma_f16(pa, bv, o[t]);
        }
    }

    // epilogue: normalize and store [B, SEQ, NPROJ] f16
#pragma unroll
    for (int rr = 0; rr < 8; ++rr) {
        float inv = 1.f / lrow[rr];
        int mm = rr + mhalf;
#pragma unroll
        for (int t = 0; t < 4; ++t) {
            O[((size_t)(b * SEQ + q0 + mm) * NPROJ) + h * HD + 16 * t + n] =
                (_Float16)(o[t][rr] * inv);
        }
    }
}

// ---------------------------------------------------------------------------
// Stage 3: output projection  Y = O @ Wo^T, written transposed as [B,C1,S].
// ---------------------------------------------------------------------------
__global__ __launch_bounds__(256)
void outproj_kernel(const _Float16* __restrict__ O,
                    const float*    __restrict__ Wo,
                    float*          __restrict__ out)
{
    const int lane = threadIdx.x & 31;
    const int wid  = blockIdx.x * 8 + (threadIdx.x >> 5);

    const int mt     = wid >> 2;        // 512 m-tiles over 8192 rows
    const int nstrip = wid & 3;         // 4 strips of 64 over C1=256
    const int row0   = mt << 4;
    const int n0     = nstrip << 6;

    const int m   = lane & 15;
    const int kb  = (lane >> 4) << 3;
    const int n   = lane & 15;
    const int kb2 = (lane >> 4) << 4;

    const _Float16* arow = O + (size_t)(row0 + m) * NPROJ;

    v8f acc[4] = {};
    for (int k0 = 0; k0 < NPROJ; k0 += 32) {
        v16h a;
        {
            v8h alo = *(const v8h*)(arow + k0 + kb2);
            v8h ahi = *(const v8h*)(arow + k0 + kb2 + 8);
#pragma unroll
            for (int i = 0; i < 8; ++i) { a[i] = alo[i]; a[i + 8] = ahi[i]; }
        }
#pragma unroll
        for (int t = 0; t < 4; ++t) {
            const float* wrow = Wo + (size_t)(n0 + 16 * t + n) * NPROJ + k0 + kb2;
            v16h bf;
#pragma unroll
            for (int i = 0; i < 16; ++i) bf[i] = (_Float16)wrow[i];
            acc[t] = wmma_f16(a, bf, acc[t]);
        }
    }

    const int b  = row0 >> 10;
    const int s0 = row0 & 1023;
#pragma unroll
    for (int t = 0; t < 4; ++t) {
#pragma unroll
        for (int rr = 0; rr < 8; ++rr) {
            int mm = rr + ((lane >> 4) << 3);
            int c  = n0 + 16 * t + n;
            out[((size_t)(b * CIN + c)) * SEQ + s0 + mm] = acc[t][rr];
        }
    }
}

// ---------------------------------------------------------------------------
extern "C" void kernel_launch(void* const* d_in, const int* in_sizes, int n_in,
                              void* d_out, int out_size, void* d_ws, size_t ws_size,
                              hipStream_t stream)
{
    (void)in_sizes; (void)n_in; (void)out_size; (void)ws_size;

    const float* input1 = (const float*)d_in[0];
    const float* input2 = (const float*)d_in[1];
    const float* Wk1    = (const float*)d_in[2];
    const float* Wk2    = (const float*)d_in[3];
    const float* Wv2    = (const float*)d_in[4];
    const float* Wo     = (const float*)d_in[5];
    float*       out    = (float*)d_out;

    const size_t NELEM = (size_t)BATCH * NHEAD * SEQ * HD;   // 4M f16 each
    _Float16* K1   = (_Float16*)d_ws;
    _Float16* K2   = K1 + NELEM;
    _Float16* V2T  = K2 + NELEM;        // [B, NH, HD, SEQ]
    _Float16* Obuf = V2T + NELEM;       // [B, SEQ, 512]

    dim3 blk(256);
    // 4096 wave-tasks / 8 waves per block = 512 blocks
    proj_gemm_kernel<false><<<512, blk, 0, stream>>>(input1, Wk1, K1);
    proj_gemm_kernel<false><<<512, blk, 0, stream>>>(input2, Wk2, K2);
    proj_gemm_kernel<true ><<<512, blk, 0, stream>>>(input2, Wv2, V2T);

    attn_kernel<<<512, blk, 0, stream>>>(K1, K2, V2T, Obuf);

    // 2048 wave-tasks / 8 = 256 blocks
    outproj_kernel<<<256, blk, 0, stream>>>(Obuf, Wo, out);
}
